// PartAtten_5145370820801
// MI455X (gfx1250) — compile-verified, matrix operand order checked by reading, not run on hardware
//
#include <hip/hip_runtime.h>

// ---------------------------------------------------------------------------
// PartAtten fused:  out = gamma * ((softmax(q k^T) @ x) @ Wv^T) + x
// q = x Wq^T, k = x Wk^T.   Key fusion: att @ (x Wv^T) == (att @ x) @ Wv^T,
// so only ONE big [24576x2048]x[2048x2048] GEMM is needed (bf16 WMMA).
// Memory-bound: ~600MB HBM for the main GEMM (Y+Wv stay L2-resident, epilogue
// streams use NT hints), ~45us-class @ 23.3 TB/s; bf16 WMMA compute trivial.
// Software-pipelined: global loads for tile t+1 overlap WMMA on tile t.
// ---------------------------------------------------------------------------

typedef __attribute__((ext_vector_type(16))) __bf16          bf16x16;
typedef __attribute__((ext_vector_type(8)))  float           f32x8;
typedef __attribute__((ext_vector_type(8)))  unsigned short  u16x8;
typedef __attribute__((ext_vector_type(4)))  unsigned short  u16x4;

union FragU { u16x8 h[2]; bf16x16 v; };

static constexpr int Bb    = 4096;
static constexpr int Ss    = 6;
static constexpr int Cc    = 2048;
static constexpr int Dd    = 256;
static constexpr int Mrows = Bb * Ss;   // 24576

#define TM  128
#define TN  128
#define TK  32
#define LDT 40   // LDS row stride (bf16 elems): 80B -> conflict-free 16B frags
#define NKT (Cc / TK)   // 64 K-tiles

__device__ __forceinline__ unsigned short f2bf_bits(float f) {
  unsigned int u = __float_as_uint(f);
  u += 0x7FFFu + ((u >> 16) & 1u);   // round-to-nearest-even
  return (unsigned short)(u >> 16);
}

// ---------------------------------------------------------------------------
// Kernel 0: Wv fp32 -> bf16 bits (4 elems/thread)
// ---------------------------------------------------------------------------
__global__ __launch_bounds__(256) void wv_to_bf16(
    const float* __restrict__ Wv, unsigned short* __restrict__ Wvb) {
  int i = (blockIdx.x * 256 + threadIdx.x) * 4;
  float4 v = *(const float4*)(Wv + i);
  u16x4 o;
  o[0] = f2bf_bits(v.x); o[1] = f2bf_bits(v.y);
  o[2] = f2bf_bits(v.z); o[3] = f2bf_bits(v.w);
  *(u16x4*)(Wvb + i) = o;
}

// ---------------------------------------------------------------------------
// Kernel 1: q/k projection GEMM.  A = x [24576 x 2048] fp32 (bf16 on stage),
// B rows = Wq (cols 0..255) or Wk (cols 256..511), row-major over K.
// Double-buffered LDS; fp32 -> bf16 conversion done in registers while the
// WMMAs for the current tile run.
// ---------------------------------------------------------------------------
__global__ __launch_bounds__(256) void qk_proj_kernel(
    const float* __restrict__ X,
    const float* __restrict__ Wq,
    const float* __restrict__ Wk,
    float* __restrict__ Qo,
    float* __restrict__ Ko) {
  __shared__ __align__(16) unsigned short sA[2][TM][LDT];
  __shared__ __align__(16) unsigned short sB[2][TN][LDT];

  const int tid  = threadIdx.x;
  const int lane = tid & 31;
  const int wid  = tid >> 5;
  const int wm   = wid & 3;        // 4 waves along M
  const int wn   = wid >> 2;       // 2 waves along N
  const int bm   = blockIdx.y * TM;
  const int bn   = blockIdx.x * TN;        // 0,128 -> Wq ; 256,384 -> Wk
  const float* Wrow = (bn < Dd) ? (Wq + (size_t)bn * Cc)
                                : (Wk + (size_t)(bn - Dd) * Cc);
  const int lrow = tid >> 1;               // 0..127
  const int lcol = (tid & 1) * 16;         // 0 or 16

  const float* gA = X    + (size_t)(bm + lrow) * Cc + lcol;
  const float* gB = Wrow + (size_t)lrow * Cc + lcol;

  f32x8 acc[2][4];
#pragma unroll
  for (int i = 0; i < 2; ++i)
#pragma unroll
    for (int j = 0; j < 4; ++j)
#pragma unroll
      for (int r = 0; r < 8; ++r) acc[i][j][r] = 0.0f;

  const int lh = lane >> 4;   // half-wave select per WMMA 16-bit layout
  const int lr = lane & 15;

  // --- prologue: stage K-tile 0 into buffer 0 ---
#pragma unroll
  for (int j = 0; j < 16; j += 4) {
    float4 a = *(const float4*)(gA + j);
    sA[0][lrow][lcol + j + 0] = f2bf_bits(a.x);
    sA[0][lrow][lcol + j + 1] = f2bf_bits(a.y);
    sA[0][lrow][lcol + j + 2] = f2bf_bits(a.z);
    sA[0][lrow][lcol + j + 3] = f2bf_bits(a.w);
    float4 b = *(const float4*)(gB + j);
    sB[0][lrow][lcol + j + 0] = f2bf_bits(b.x);
    sB[0][lrow][lcol + j + 1] = f2bf_bits(b.y);
    sB[0][lrow][lcol + j + 2] = f2bf_bits(b.z);
    sB[0][lrow][lcol + j + 3] = f2bf_bits(b.w);
  }
  __syncthreads();

  for (int kt = 0; kt < NKT; ++kt) {
    const int cur = kt & 1, nxt = cur ^ 1;
    const bool more = (kt + 1) < NKT;

    // issue next tile's global loads first (latency hidden behind WMMAs)
    float4 ra[4], rb[4];
    if (more) {
      const float* nA = gA + (kt + 1) * TK;
      const float* nB = gB + (kt + 1) * TK;
#pragma unroll
      for (int j = 0; j < 4; ++j) {
        ra[j] = *(const float4*)(nA + 4 * j);
        rb[j] = *(const float4*)(nB + 4 * j);
      }
      if (kt + 2 < NKT) {
        __builtin_prefetch(gA + (kt + 2) * TK, 0, 1);  // global_prefetch_b8
        __builtin_prefetch(gB + (kt + 2) * TK, 0, 1);
      }
    }

    // compute from current buffer
    FragU a[2], b[4];
#pragma unroll
    for (int i = 0; i < 2; ++i) {
      int r = 32 * wm + 16 * i + lr;
      a[i].h[0] = *(const u16x8*)&sA[cur][r][lh * 8];
      a[i].h[1] = *(const u16x8*)&sA[cur][r][16 + lh * 8];
    }
#pragma unroll
    for (int j = 0; j < 4; ++j) {
      int c = 64 * wn + 16 * j + lr;
      b[j].h[0] = *(const u16x8*)&sB[cur][c][lh * 8];
      b[j].h[1] = *(const u16x8*)&sB[cur][c][16 + lh * 8];
    }
#pragma unroll
    for (int i = 0; i < 2; ++i)
#pragma unroll
      for (int j = 0; j < 4; ++j)
        acc[i][j] = __builtin_amdgcn_wmma_f32_16x16x32_bf16(
            false, a[i].v, false, b[j].v, (short)0, acc[i][j], false, false);

    // convert + spill staged registers into the other buffer
    if (more) {
#pragma unroll
      for (int j = 0; j < 4; ++j) {
        sA[nxt][lrow][lcol + 4 * j + 0] = f2bf_bits(ra[j].x);
        sA[nxt][lrow][lcol + 4 * j + 1] = f2bf_bits(ra[j].y);
        sA[nxt][lrow][lcol + 4 * j + 2] = f2bf_bits(ra[j].z);
        sA[nxt][lrow][lcol + 4 * j + 3] = f2bf_bits(ra[j].w);
        sB[nxt][lrow][lcol + 4 * j + 0] = f2bf_bits(rb[j].x);
        sB[nxt][lrow][lcol + 4 * j + 1] = f2bf_bits(rb[j].y);
        sB[nxt][lrow][lcol + 4 * j + 2] = f2bf_bits(rb[j].z);
        sB[nxt][lrow][lcol + 4 * j + 3] = f2bf_bits(rb[j].w);
      }
    }
    __syncthreads();
  }

  // epilogue: fp32 write, split into Q / K halves (block-uniform)
  const int ln = lane & 15;
#pragma unroll
  for (int i = 0; i < 2; ++i)
#pragma unroll
    for (int j = 0; j < 4; ++j) {
      int gmB = bm + 32 * wm + 16 * i;
      int gn  = bn + 64 * wn + 16 * j + ln;
#pragma unroll
      for (int r = 0; r < 8; ++r) {
        int gm = gmB + r + 8 * lh;
        float vv = acc[i][j][r];
        if (gn < Dd) Qo[(size_t)gm * Dd + gn]        = vv;
        else         Ko[(size_t)gm * Dd + (gn - Dd)] = vv;
      }
    }
}

// ---------------------------------------------------------------------------
// Kernel 2: per-batch attention (6x6) + y = att @ x (bf16 out), att -> d_out
// ---------------------------------------------------------------------------
__global__ __launch_bounds__(256) void attn_kernel(
    const float* __restrict__ Q,
    const float* __restrict__ Kb,
    const float* __restrict__ X,
    unsigned short* __restrict__ Y,
    float* __restrict__ AttOut) {
  const int b   = blockIdx.x;
  const int tid = threadIdx.x;
  __shared__ float sq[Ss][Dd];
  __shared__ float sk[Ss][Dd];
  __shared__ float satt[Ss][Ss];

#pragma unroll
  for (int s = 0; s < Ss; ++s) {
    sq[s][tid] = Q [((size_t)b * Ss + s) * Dd + tid];
    sk[s][tid] = Kb[((size_t)b * Ss + s) * Dd + tid];
  }
  __syncthreads();

  if (tid < Ss * Ss) {
    int s = tid / Ss, t = tid % Ss;
    float e = 0.f;
    for (int d = 0; d < Dd; ++d) e += sq[s][d] * sk[t][d];
    satt[s][t] = e;
  }
  __syncthreads();

  if (tid < Ss) {  // softmax over keys, row tid
    float mx = satt[tid][0];
#pragma unroll
    for (int t = 1; t < Ss; ++t) mx = fmaxf(mx, satt[tid][t]);
    float e[Ss], sum = 0.f;
#pragma unroll
    for (int t = 0; t < Ss; ++t) { e[t] = __expf(satt[tid][t] - mx); sum += e[t]; }
    float inv = 1.f / sum;
#pragma unroll
    for (int t = 0; t < Ss; ++t) satt[tid][t] = e[t] * inv;
  }
  __syncthreads();

  if (tid < Ss * Ss)
    AttOut[(size_t)b * Ss * Ss + tid] = satt[tid / Ss][tid % Ss];

  // y[s][c] = sum_t att[s][t] * x[b][t][c]   (bf16 out for the big GEMM)
  for (int c = tid; c < Cc; c += 256) {
    float xv[Ss];
#pragma unroll
    for (int t = 0; t < Ss; ++t) xv[t] = X[((size_t)b * Ss + t) * Cc + c];
#pragma unroll
    for (int s = 0; s < Ss; ++s) {
      float a = 0.f;
#pragma unroll
      for (int t = 0; t < Ss; ++t) a += satt[s][t] * xv[t];
      Y[((size_t)b * Ss + s) * Cc + c] = f2bf_bits(a);
    }
  }
}

// ---------------------------------------------------------------------------
// Kernel 3: out = gamma * (y @ Wv^T) + x   (bf16 WMMA GEMM, fp32 accum)
// A = y [24576 x 2048] bf16, B rows = Wv [2048 x 2048] bf16 (row-major over K)
// Double-buffered LDS; epilogue uses NT load/store so the 400MB residual/out
// streams do not evict the L2-resident Y (100MB) + Wv (8MB) operands.
// ---------------------------------------------------------------------------
__global__ __launch_bounds__(256) void gemm_out_kernel(
    const unsigned short* __restrict__ Y,
    const unsigned short* __restrict__ Wvb,
    const float* __restrict__ Xres,
    const float* __restrict__ gamma,
    float* __restrict__ Out) {
  __shared__ __align__(16) unsigned short sA[2][TM][LDT];
  __shared__ __align__(16) unsigned short sB[2][TN][LDT];

  const int tid  = threadIdx.x;
  const int lane = tid & 31;
  const int wid  = tid >> 5;
  const int wm   = wid & 3;
  const int wn   = wid >> 2;
  const int bm   = blockIdx.y * TM;
  const int bn   = blockIdx.x * TN;
  const int lrow = tid >> 1;
  const int lcol = (tid & 1) * 16;

  const unsigned short* gA = Y   + (size_t)(bm + lrow) * Cc + lcol;
  const unsigned short* gB = Wvb + (size_t)(bn + lrow) * Cc + lcol;

  f32x8 acc[2][4];
#pragma unroll
  for (int i = 0; i < 2; ++i)
#pragma unroll
    for (int j = 0; j < 4; ++j)
#pragma unroll
      for (int r = 0; r < 8; ++r) acc[i][j][r] = 0.0f;

  const int lh = lane >> 4;
  const int lr = lane & 15;

  // --- prologue: stage K-tile 0 into buffer 0 ---
  *(u16x8*)&sA[0][lrow][lcol]     = *(const u16x8*)(gA);
  *(u16x8*)&sA[0][lrow][lcol + 8] = *(const u16x8*)(gA + 8);
  *(u16x8*)&sB[0][lrow][lcol]     = *(const u16x8*)(gB);
  *(u16x8*)&sB[0][lrow][lcol + 8] = *(const u16x8*)(gB + 8);
  __syncthreads();

  for (int kt = 0; kt < NKT; ++kt) {
    const int cur = kt & 1, nxt = cur ^ 1;
    const bool more = (kt + 1) < NKT;

    // issue next tile's global loads (hidden behind WMMAs)
    u16x8 ra0, ra1, rb0, rb1;
    if (more) {
      const unsigned short* nA = gA + (kt + 1) * TK;
      const unsigned short* nB = gB + (kt + 1) * TK;
      ra0 = *(const u16x8*)(nA);
      ra1 = *(const u16x8*)(nA + 8);
      rb0 = *(const u16x8*)(nB);
      rb1 = *(const u16x8*)(nB + 8);
      if (kt + 2 < NKT) {
        __builtin_prefetch(gA + (kt + 2) * TK, 0, 1);  // global_prefetch_b8
        __builtin_prefetch(gB + (kt + 2) * TK, 0, 1);
      }
    }

    // compute from current buffer
    FragU a[2], b[4];
#pragma unroll
    for (int i = 0; i < 2; ++i) {
      int r = 32 * wm + 16 * i + lr;
      a[i].h[0] = *(const u16x8*)&sA[cur][r][lh * 8];
      a[i].h[1] = *(const u16x8*)&sA[cur][r][16 + lh * 8];
    }
#pragma unroll
    for (int j = 0; j < 4; ++j) {
      int c = 64 * wn + 16 * j + lr;
      b[j].h[0] = *(const u16x8*)&sB[cur][c][lh * 8];
      b[j].h[1] = *(const u16x8*)&sB[cur][c][16 + lh * 8];
    }
#pragma unroll
    for (int i = 0; i < 2; ++i)
#pragma unroll
      for (int j = 0; j < 4; ++j)
        acc[i][j] = __builtin_amdgcn_wmma_f32_16x16x32_bf16(
            false, a[i].v, false, b[j].v, (short)0, acc[i][j], false, false);

    // spill staged registers into the other buffer
    if (more) {
      *(u16x8*)&sA[nxt][lrow][lcol]     = ra0;
      *(u16x8*)&sA[nxt][lrow][lcol + 8] = ra1;
      *(u16x8*)&sB[nxt][lrow][lcol]     = rb0;
      *(u16x8*)&sB[nxt][lrow][lcol + 8] = rb1;
    }
    __syncthreads();
  }

  const float g  = gamma[0];
  const int   ln = lane & 15;
#pragma unroll
  for (int i = 0; i < 2; ++i)
#pragma unroll
    for (int j = 0; j < 4; ++j) {
      int gmB = bm + 32 * wm + 16 * i;
      int gn  = bn + 64 * wn + 16 * j + ln;
#pragma unroll
      for (int r = 0; r < 8; ++r) {
        int gm = gmB + r + 8 * lh;
        size_t idx = (size_t)gm * Cc + gn;
        float xr = __builtin_nontemporal_load(&Xres[idx]);      // NT: don't pollute L2
        __builtin_nontemporal_store(g * acc[i][j][r] + xr, &Out[idx]);
      }
    }
}

// ---------------------------------------------------------------------------
extern "C" void kernel_launch(void* const* d_in, const int* in_sizes, int n_in,
                              void* d_out, int out_size, void* d_ws, size_t ws_size,
                              hipStream_t stream) {
  const float* x     = (const float*)d_in[0];
  const float* Wq    = (const float*)d_in[1];
  const float* Wk    = (const float*)d_in[2];
  const float* Wv    = (const float*)d_in[3];
  const float* gamma = (const float*)d_in[4];

  float* out    = (float*)d_out;
  float* attOut = out + (size_t)Mrows * Cc;   // tuple: (out, attention)

  // workspace layout (bytes):
  //   q    fp32  [24576 x 256]   25,165,824
  //   k    fp32  [24576 x 256]   25,165,824
  //   y    bf16  [24576 x 2048] 100,663,296
  //   Wvb  bf16  [2048 x 2048]    8,388,608   (total ~152 MB)
  char* ws = (char*)d_ws;
  float*          q   = (float*)(ws);
  float*          k   = (float*)(ws + 25165824);
  unsigned short* y   = (unsigned short*)(ws + 50331648);
  unsigned short* wvb = (unsigned short*)(ws + 150994944);

  wv_to_bf16<<<(Cc * Cc) / (256 * 4), 256, 0, stream>>>(Wv, wvb);
  qk_proj_kernel<<<dim3((2 * Dd) / TN, Mrows / TM), 256, 0, stream>>>(x, Wq, Wk, q, k);
  attn_kernel<<<Bb, 256, 0, stream>>>(q, k, x, y, attOut);
  gemm_out_kernel<<<dim3(Cc / TN, Mrows / TM), 256, 0, stream>>>(y, wvb, x, gamma, out);
}
